// LSTM_10634339025607
// MI455X (gfx1250) — matrix operand order repaired and checked by reading, not compiled, not run on hardware
//
#include <hip/hip_runtime.h>
#include <hip/hip_bf16.h>
#include <stdint.h>

#define B_   256
#define T_   512
#define IN_  64
#define H_   128
#define G_   512   // 4*H
#define NWG  16
#define WAVES_PER_WG 8

#define LDS_FRAG_STRIDE  1024   // bytes per B fragment (32 lanes * 32B)
#define LDS_CHUNK_STRIDE 4096   // 4 gate fragments per k-chunk

typedef __attribute__((ext_vector_type(16))) __bf16 bf16x16;
typedef __attribute__((ext_vector_type(8)))  float  floatx8;

// ---------- helpers ----------

static __device__ __forceinline__ unsigned short f32_to_bf16_rne(float f) {
  uint32_t u = __float_as_uint(f);
  uint32_t r = u + 0x7FFFu + ((u >> 16) & 1u);
  return (unsigned short)(r >> 16);
}

// A-matrix fragment (16x32 bf16, row-major source, ld = row stride in elems).
// ISA layout: lanes 0-15: M=lane, VGPR0-3 K=0..7 (pairs), VGPR4-7 K=16..23.
//             lanes 16-31: M=lane-16, VGPR0-3 K=8..15, VGPR4-7 K=24..31.
static __device__ __forceinline__ bf16x16 load_a_frag(const __bf16* base, int ld) {
  int l  = threadIdx.x & 31;
  int m  = l & 15;
  int kb = (l < 16) ? 0 : 8;
  const __bf16* p = base + (size_t)m * ld + kb;
  union { bf16x16 v; uint4 q[2]; } u;
  u.q[0] = *(const uint4*)(p);        // K = kb .. kb+7
  u.q[1] = *(const uint4*)(p + 16);   // K = kb+16 .. kb+23
  return u.v;
}

// Load the 4 gate B-fragments of one k-chunk from LDS (pre-swizzled so each
// lane's 32 bytes are contiguous at addr + gate*1024). asm volatile prevents
// LICM from hoisting these loop-invariant loads into (spilled) registers.
static __device__ __forceinline__ void lds_load_bfrag4(unsigned addr, bf16x16 out[4]) {
  uint4 a0, a1, b0, b1, c0, c1, d0, d1;
  asm volatile(
      "ds_load_b128 %0, %8 offset:0\n\t"
      "ds_load_b128 %1, %8 offset:16\n\t"
      "ds_load_b128 %2, %8 offset:1024\n\t"
      "ds_load_b128 %3, %8 offset:1040\n\t"
      "ds_load_b128 %4, %8 offset:2048\n\t"
      "ds_load_b128 %5, %8 offset:2064\n\t"
      "ds_load_b128 %6, %8 offset:3072\n\t"
      "ds_load_b128 %7, %8 offset:3088\n\t"
      "s_wait_dscnt 0x0"
      : "=&v"(a0), "=&v"(a1), "=&v"(b0), "=&v"(b1),
        "=&v"(c0), "=&v"(c1), "=&v"(d0), "=&v"(d1)
      : "v"(addr));
  union { bf16x16 v; uint4 q[2]; } u;
  u.q[0] = a0; u.q[1] = a1; out[0] = u.v;
  u.q[0] = b0; u.q[1] = b1; out[1] = u.v;
  u.q[0] = c0; u.q[1] = c1; out[2] = u.v;
  u.q[0] = d0; u.q[1] = d1; out[3] = u.v;
}

// Sense-free grid barrier via monotonically increasing generation counter.
// bar[0] = arrival counter, bar[1] = generation. Zeroed before launch.
static __device__ __forceinline__ void grid_barrier(unsigned* bar, unsigned target) {
  __threadfence();                 // make this thread's stores visible
  __syncthreads();                 // all waves of WG done + fenced
  if (threadIdx.x == 0) {
    unsigned prev = atomicAdd(&bar[0], 1u);
    if (prev == NWG - 1) {
      __hip_atomic_store(&bar[0], 0u, __ATOMIC_RELAXED, __HIP_MEMORY_SCOPE_AGENT);
      __threadfence();
      atomicAdd(&bar[1], 1u);
    } else {
      while (__hip_atomic_load(&bar[1], __ATOMIC_RELAXED, __HIP_MEMORY_SCOPE_AGENT) < target) {
        __builtin_amdgcn_s_sleep(1);
      }
    }
    __threadfence();
  }
  __syncthreads();
}

// ---------- fused persistent LSTM layer ----------
// gates(t) = [ x_t | h_{t-1} ] * [ W_ih^T | W_hh^T ] + bias, then pointwise.
// 128 waves: wave = 16 batch-row tiles x 8 H-column groups (16 cols each).
// Each wave owns the 4 gate tiles (i,f,g,o) of its strip -> lane-local update.
// Weights live in LDS, pre-swizzled into WMMA fragment layout.
template<int K_IN>
__global__ __launch_bounds__(256) void lstm_layer_kernel(
    const __bf16* __restrict__ xin, long x_tstride, long x_rstride,
    const __bf16* __restrict__ w_ih,   // [4H, K_IN] bf16
    const __bf16* __restrict__ w_hh,   // [4H, H]    bf16
    const float*  __restrict__ bias,   // [4H] combined b_ih+b_hh
    __bf16* __restrict__ hbuf0,        // ping-pong [B*H] bf16 (zeroed)
    __bf16* __restrict__ hbuf1,
    unsigned short* __restrict__ hist, // [T,B,H] bf16 history (or null)
    float* __restrict__ hlast,         // [B,H] f32 final h (or null)
    unsigned* __restrict__ bar)
{
  extern __shared__ char smem[];
  constexpr int NCX = K_IN / 32;       // x-side k-chunks (2 or 4)
  constexpr int NCH = H_ / 32;         // h-side k-chunks (4)
  constexpr int NCHUNK = NCX + NCH;    // 6 or 8

  const int wave   = threadIdx.x >> 5;
  const int wid    = blockIdx.x * WAVES_PER_WG + wave;  // 0..127
  const int row0   = (wid >> 3) * 16;                   // batch tile base
  const int hcol0  = (wid & 7) * 16;                    // H-column strip base
  const int l      = threadIdx.x & 31;
  const int n      = l & 15;                            // D column within tile
  const int mstart = (l >> 4) * 8;                      // D row base within tile

  // ---- stage this wave's weight fragments into LDS (fragment-lane layout) ----
  char* wbase = smem + (size_t)wave * (NCHUNK * LDS_CHUNK_STRIDE);
  {
    const int kb = (l < 16) ? 0 : 8;    // B-frag: lanes 0-15 K=0..7,16..23; 16-31 K=8..15,24..31
    #pragma unroll
    for (int c = 0; c < NCHUNK; ++c) {
      const __bf16* src; int ld, k;
      if (c < NCX) { src = w_ih; ld = K_IN; k = c * 32; }
      else         { src = w_hh; ld = H_;   k = (c - NCX) * 32; }
      #pragma unroll
      for (int q = 0; q < 4; ++q) {
        const __bf16* p = src + (size_t)(q * H_ + hcol0 + n) * ld + k + kb;
        uint4 lo = *(const uint4*)(p);
        uint4 hi = *(const uint4*)(p + 16);
        char* dst = wbase + c * LDS_CHUNK_STRIDE + q * LDS_FRAG_STRIDE + l * 32;
        *(uint4*)(dst)      = lo;
        *(uint4*)(dst + 16) = hi;
      }
    }
  }
  __syncthreads();
  const unsigned lds_lane_base = (unsigned)(uintptr_t)wbase + (unsigned)(l * 32);

  float c_reg[8];
  #pragma unroll
  for (int r = 0; r < 8; ++r) c_reg[r] = 0.f;

  float bias_g[4];
  #pragma unroll
  for (int q = 0; q < 4; ++q) bias_g[q] = bias[q * H_ + hcol0 + n];

  unsigned target = 0;

  for (int t = 0; t < T_; ++t) {
    const __bf16* hread  = (t & 1) ? hbuf1 : hbuf0;
    __bf16*       hwrite = (t & 1) ? hbuf0 : hbuf1;

    floatx8 acc[4] = {};

    // input-projection contribution: x_t tile [16 x K_IN]
    const __bf16* xbase = xin + (size_t)t * x_tstride + (size_t)row0 * x_rstride;
    #pragma unroll
    for (int c = 0; c < NCX; ++c) {
      bf16x16 a = load_a_frag(xbase + c * 32, (int)x_rstride);
      bf16x16 bq[4];
      lds_load_bfrag4(lds_lane_base + c * LDS_CHUNK_STRIDE, bq);
      #pragma unroll
      for (int q = 0; q < 4; ++q)
        acc[q] = __builtin_amdgcn_wmma_f32_16x16x32_bf16(
            false, a, false, bq[q], (short)0, acc[q], false, false);
    }
    // recurrent contribution: h_{t-1} tile [16 x H]
    const __bf16* hbase = hread + (size_t)row0 * H_;
    #pragma unroll
    for (int c = 0; c < NCH; ++c) {
      bf16x16 a = load_a_frag(hbase + c * 32, H_);
      bf16x16 bq[4];
      lds_load_bfrag4(lds_lane_base + (NCX + c) * LDS_CHUNK_STRIDE, bq);
      #pragma unroll
      for (int q = 0; q < 4; ++q)
        acc[q] = __builtin_amdgcn_wmma_f32_16x16x32_bf16(
            false, a, false, bq[q], (short)0, acc[q], false, false);
    }

    // pointwise LSTM cell update (all four gates lane-local, f32)
    #pragma unroll
    for (int r = 0; r < 8; ++r) {
      int row = row0 + mstart + r;
      int col = hcol0 + n;
      float pi = acc[0][r] + bias_g[0];
      float pf = acc[1][r] + bias_g[1];
      float pg = acc[2][r] + bias_g[2];
      float po = acc[3][r] + bias_g[3];
      float gi = 1.f / (1.f + __expf(-pi));
      float gf = 1.f / (1.f + __expf(-pf));
      float gg = tanhf(pg);
      float go = 1.f / (1.f + __expf(-po));
      float c  = gf * c_reg[r] + gi * gg;
      c_reg[r] = c;
      float h  = go * tanhf(c);
      unsigned short hb = f32_to_bf16_rne(h);
      ((unsigned short*)hwrite)[row * H_ + col] = hb;
      if (hist)  hist[((size_t)t * B_ + row) * H_ + col] = hb;
      if (hlast && t == T_ - 1) hlast[row * H_ + col] = h;
    }

    ++target;
    grid_barrier(bar, target);
  }
}

// ---------- small utility kernels ----------

__global__ void f32_to_bf16_kernel(const float* __restrict__ in,
                                   unsigned short* __restrict__ out, size_t n) {
  size_t i = (size_t)blockIdx.x * blockDim.x + threadIdx.x;
  size_t stride = (size_t)gridDim.x * blockDim.x;
  for (; i < n; i += stride) out[i] = f32_to_bf16_rne(in[i]);
}

__global__ void bias_combine_kernel(const float* __restrict__ a,
                                    const float* __restrict__ b,
                                    float* __restrict__ out, int n) {
  int i = blockIdx.x * blockDim.x + threadIdx.x;
  if (i < n) out[i] = a[i] + b[i];
}

__global__ void fc_kernel(const float* __restrict__ hlast,
                          const float* __restrict__ fcw,
                          const float* __restrict__ fcb,
                          float* __restrict__ out) {
  int b = blockIdx.x * blockDim.x + threadIdx.x;
  if (b < B_) {
    float s = 0.f;
    #pragma unroll
    for (int k = 0; k < H_; ++k) s += hlast[b * H_ + k] * fcw[k];
    out[b] = s + fcb[0];
  }
}

// ---------- host ----------

extern "C" void kernel_launch(void* const* d_in, const int* in_sizes, int n_in,
                              void* d_out, int out_size, void* d_ws, size_t ws_size,
                              hipStream_t stream) {
  const float* x     = (const float*)d_in[0];
  const float* w_ih0 = (const float*)d_in[1];
  const float* w_hh0 = (const float*)d_in[2];
  const float* b_ih0 = (const float*)d_in[3];
  const float* b_hh0 = (const float*)d_in[4];
  const float* w_ih1 = (const float*)d_in[5];
  const float* w_hh1 = (const float*)d_in[6];
  const float* b_ih1 = (const float*)d_in[7];
  const float* b_hh1 = (const float*)d_in[8];
  const float* fc_w  = (const float*)d_in[9];
  const float* fc_b  = (const float*)d_in[10];
  float* out = (float*)d_out;

  char* ws = (char*)d_ws;
  size_t off = 0;
  auto alloc = [&](size_t bytes) -> char* {
    off = (off + 255) & ~(size_t)255;
    char* p = ws + off;
    off += bytes;
    return p;
  };

  unsigned*       bar    = (unsigned*)alloc(256);
  unsigned short* hbuf   = (unsigned short*)alloc((size_t)2 * B_ * H_ * 2); // ping-pong
  unsigned short* xbf    = (unsigned short*)alloc((size_t)B_ * T_ * IN_ * 2);
  unsigned short* wih0b  = (unsigned short*)alloc((size_t)G_ * IN_ * 2);
  unsigned short* whh0b  = (unsigned short*)alloc((size_t)G_ * H_ * 2);
  unsigned short* wih1b  = (unsigned short*)alloc((size_t)G_ * H_ * 2);
  unsigned short* whh1b  = (unsigned short*)alloc((size_t)G_ * H_ * 2);
  float*          bias0  = (float*)alloc((size_t)G_ * 4);
  float*          bias1  = (float*)alloc((size_t)G_ * 4);
  unsigned short* h0hist = (unsigned short*)alloc((size_t)T_ * B_ * H_ * 2);
  float*          hlast  = (float*)alloc((size_t)B_ * H_ * 4);
  (void)ws_size; (void)in_sizes; (void)n_in; (void)out_size;

  // init barrier + ping-pong (h_{-1} = 0)
  hipMemsetAsync(bar, 0, 256, stream);
  hipMemsetAsync(hbuf, 0, (size_t)2 * B_ * H_ * 2, stream);

  // convert inputs/weights to bf16; combine biases
  f32_to_bf16_kernel<<<2048, 256, 0, stream>>>(x, xbf, (size_t)B_ * T_ * IN_);
  f32_to_bf16_kernel<<<64, 256, 0, stream>>>(w_ih0, wih0b, (size_t)G_ * IN_);
  f32_to_bf16_kernel<<<64, 256, 0, stream>>>(w_hh0, whh0b, (size_t)G_ * H_);
  f32_to_bf16_kernel<<<64, 256, 0, stream>>>(w_ih1, wih1b, (size_t)G_ * H_);
  f32_to_bf16_kernel<<<64, 256, 0, stream>>>(w_hh1, whh1b, (size_t)G_ * H_);
  bias_combine_kernel<<<2, 256, 0, stream>>>(b_ih0, b_hh0, bias0, G_);
  bias_combine_kernel<<<2, 256, 0, stream>>>(b_ih1, b_hh1, bias1, G_);

  // layer 0: x is [B,T,IN] -> t-stride = IN, row(b)-stride = T*IN
  // LDS: 8 waves * 6 chunks * 4KB = 192KB
  lstm_layer_kernel<IN_><<<NWG, 256, (IN_/32 + H_/32) * LDS_CHUNK_STRIDE * WAVES_PER_WG, stream>>>(
      (const __bf16*)xbf, (long)IN_, (long)(T_ * IN_),
      (const __bf16*)wih0b, (const __bf16*)whh0b, bias0,
      (__bf16*)hbuf, (__bf16*)(hbuf + (size_t)B_ * H_),
      h0hist, nullptr, bar);

  hipMemsetAsync(bar, 0, 256, stream);
  hipMemsetAsync(hbuf, 0, (size_t)2 * B_ * H_ * 2, stream);

  // layer 1: input is h0 history [T,B,H] -> t-stride = B*H, row-stride = H
  // LDS: 8 waves * 8 chunks * 4KB = 256KB
  lstm_layer_kernel<H_><<<NWG, 256, (H_/32 + H_/32) * LDS_CHUNK_STRIDE * WAVES_PER_WG, stream>>>(
      (const __bf16*)h0hist, (long)(B_ * H_), (long)H_,
      (const __bf16*)wih1b, (const __bf16*)whh1b, bias1,
      (__bf16*)hbuf, (__bf16*)(hbuf + (size_t)B_ * H_),
      nullptr, hlast, bar);

  // final FC: [B,H] x [H] + b -> [B,1]
  fc_kernel<<<1, 256, 0, stream>>>(hlast, fc_w, fc_b, out);
}
